// GCNLinkPredictShare_76544907149488
// MI455X (gfx1250) — matrix operand order repaired and checked by reading, not compiled, not run on hardware
//
#include <hip/hip_runtime.h>

typedef __attribute__((ext_vector_type(16))) _Float16 v16h;
typedef __attribute__((ext_vector_type(8)))  float    v8f;

// ---------------------------------------------------------------------------
// Weight pre-pass: W[K,Nc] f32  ->  Wt[Nc,K] f16 (transposed, 32B-aligned rows)
// so each lane's B fragment is a single contiguous 32-byte v16h load.
// ---------------------------------------------------------------------------
__global__ void k_w2h_t(const float* __restrict__ W, _Float16* __restrict__ Wt,
                        int K, int Nc) {
    int gid = blockIdx.x * blockDim.x + threadIdx.x;
    if (gid >= K * Nc) return;
    int n = gid / K, k = gid - n * K;
    Wt[gid] = (_Float16)W[(long)k * Nc + n];
}

// ---------------------------------------------------------------------------
// WMMA GEMM: C[M, Nc] (op)= relu?( A'[M,K] @ W[K,Nc] + bias )
//   A'[m,k] = A[m*lda+k] * tf(m); TMODE 0: tf=1-types[m], 1: tf=types[m], 2: 1
//   ADD: C += result.  HB: bias present.  sidx swap for small_idx==2.
// One wave per 16x64 output tile: one A fragment + 4 B fragments (hoisted so
// loads overlap the WMMA chain) -> 4 v_wmma_f32_16x16x32_f16 per k-step.
// Requires M % 16 == 0 (M = 100000 = 6250*16) and Nc % 64 == 0.
// Fragment layouts used here (same contiguous k order in A and B):
//   A 16x32 f16 : lane m = lane%16; pair p holds K=half*16 + 2p
//                 -> one contiguous 64B f32 span per lane, loaded as 4x float4
//   B 32x16 f16 : lane n = lane%16; VGPR v holds K = half*16 + 2v (contig in Wt)
//   C/D 16x16   : VGPR r holds row half*8+r, col = lane%16
// ---------------------------------------------------------------------------
template<int TMODE, int RELU, int ADD, int HB>
__global__ __launch_bounds__(32)
void wmma_gemm(const float* __restrict__ A, int lda,
               const _Float16* __restrict__ Wt, const float* __restrict__ bias,
               const _Float16* __restrict__ Wt2, const float* __restrict__ bias2,
               const int* __restrict__ sidx,
               const float* __restrict__ types,
               float* __restrict__ C, int M, int K, int Nc)
{
    if (sidx && *sidx == 2) { Wt = Wt2; bias = bias2; }

    const int lane    = threadIdx.x & 31;
    const int half_id = lane >> 4;
    const int mr      = lane & 15;
    const int tile_m  = blockIdx.x;
    const int cbase   = blockIdx.y * 64;        // 4 n-tiles of 16

    const int arow = tile_m * 16 + mr;
    float tf = 1.0f;
    if (TMODE == 0) tf = 1.0f - types[arow];
    if (TMODE == 1) tf = types[arow];

    const float*    ap = A  + (long)arow * lda + (half_id << 4);
    const _Float16* wp = Wt + (long)(cbase + mr) * K + (half_id << 4);

    v8f acc[4] = {{}, {}, {}, {}};
    for (int k0 = 0; k0 < K; k0 += 32) {
        // ---- 4 B fragments first: staggered loadcnt waits, overlap with WMMAs
        v16h b[4];
#pragma unroll
        for (int j = 0; j < 4; ++j)
            b[j] = *(const v16h*)(wp + (long)j * 16 * K + k0);     // 32B load

        // ---- A fragment: one contiguous span K = half*16 + 0..15 -> 4x global_load_b128
        float4 c0 = *(const float4*)(ap + k0);
        float4 c1 = *(const float4*)(ap + k0 + 4);
        float4 c2 = *(const float4*)(ap + k0 + 8);
        float4 c3 = *(const float4*)(ap + k0 + 12);
        v16h a;
        a[0]  = (_Float16)(c0.x * tf); a[1]  = (_Float16)(c0.y * tf);
        a[2]  = (_Float16)(c0.z * tf); a[3]  = (_Float16)(c0.w * tf);
        a[4]  = (_Float16)(c1.x * tf); a[5]  = (_Float16)(c1.y * tf);
        a[6]  = (_Float16)(c1.z * tf); a[7]  = (_Float16)(c1.w * tf);
        a[8]  = (_Float16)(c2.x * tf); a[9]  = (_Float16)(c2.y * tf);
        a[10] = (_Float16)(c2.z * tf); a[11] = (_Float16)(c2.w * tf);
        a[12] = (_Float16)(c3.x * tf); a[13] = (_Float16)(c3.y * tf);
        a[14] = (_Float16)(c3.z * tf); a[15] = (_Float16)(c3.w * tf);

#pragma unroll
        for (int j = 0; j < 4; ++j)
            acc[j] = __builtin_amdgcn_wmma_f32_16x16x32_f16(
                         false, a, false, b[j], (short)0, acc[j], false, false);
    }

    const int grow0 = tile_m * 16 + (half_id << 3);
#pragma unroll
    for (int j = 0; j < 4; ++j) {
        const int gcol = cbase + j * 16 + mr;
        const float bval = HB ? bias[gcol] : 0.0f;
#pragma unroll
        for (int r = 0; r < 8; ++r) {
            long idx = (long)(grow0 + r) * Nc + gcol;
            float v = acc[j][r] + bval;
            if (ADD)  v += C[idx];
            if (RELU) v = fmaxf(v, 0.0f);
            C[idx] = v;
        }
    }
}

// ---------------------------------------------------------------------------
// GCN support kernels (bandwidth/atomic-bound part)
// ---------------------------------------------------------------------------
__global__ void k_fill_deg(float* __restrict__ deg, int n) {
    int i = blockIdx.x * blockDim.x + threadIdx.x;
    if (i < n) deg[i] = 1.0f;                      // self-loop
}

__global__ void k_edge_deg(const int* __restrict__ col, float* __restrict__ deg, int E) {
    int e = blockIdx.x * blockDim.x + threadIdx.x;
    if (e < E) atomicAdd(&deg[col[e]], 1.0f);
}

__global__ void k_rsqrt(const float* __restrict__ deg, float* __restrict__ dis, int n) {
    int i = blockIdx.x * blockDim.x + threadIdx.x;
    if (i < n) dis[i] = rsqrtf(deg[i]);
}

// S[i,f] = G[i,f] * dis[i]^2   (self-loop term doubles as scatter-buffer init)
__global__ void k_selfinit(const float* __restrict__ G, const float* __restrict__ dis,
                           float* __restrict__ S, long total) {
    long gid = (long)blockIdx.x * blockDim.x + threadIdx.x;
    if (gid >= total) return;
    float d = dis[gid >> 7];                       // F = 128
    S[gid] = G[gid] * d * d;
}

// S[col[e],f] += G[row[e],f] * dis[row]*dis[col]; coalesced across f
__global__ void k_edge_scatter(const int* __restrict__ row, const int* __restrict__ col,
                               const float* __restrict__ G, const float* __restrict__ dis,
                               float* __restrict__ S, long total) {
    long gid = (long)blockIdx.x * blockDim.x + threadIdx.x;
    if (gid >= total) return;
    long e = gid >> 7;
    int  f = (int)(gid & 127);
    int  r = row[e], c = col[e];
    float w = dis[r] * dis[c];
    atomicAdd(&S[(long)c * 128 + f], G[(long)r * 128 + f] * w);
}

// X[i,f] = relu(X[i,f] + b[f]),  F = 128
__global__ void k_bias_relu(float* __restrict__ X, const float* __restrict__ b, long total) {
    long gid = (long)blockIdx.x * blockDim.x + threadIdx.x;
    if (gid >= total) return;
    X[gid] = fmaxf(X[gid] + b[gid & 127], 0.0f);
}

__global__ void k_add(float* __restrict__ A, const float* __restrict__ B, long n) {
    long gid = (long)blockIdx.x * blockDim.x + threadIdx.x;
    if (gid < n) A[gid] += B[gid];
}

// ---------------------------------------------------------------------------
extern "C" void kernel_launch(void* const* d_in, const int* in_sizes, int n_in,
                              void* d_out, int out_size, void* d_ws, size_t ws_size,
                              hipStream_t stream) {
    const float* x       = (const float*)d_in[0];
    const float* types   = (const float*)d_in[1];
    const int*   ei      = (const int*)d_in[2];
    const int*   sidx    = (const int*)d_in[3];
    const float* hete0_W = (const float*)d_in[4];
    const float* hete0_b = (const float*)d_in[5];
    const float* hete1_W = (const float*)d_in[6];
    const float* hete1_b = (const float*)d_in[7];
    const float* a0_W1   = (const float*)d_in[8];
    const float* a0_b1   = (const float*)d_in[9];
    const float* a0_W2   = (const float*)d_in[10];
    const float* a0_b2   = (const float*)d_in[11];
    const float* a1_W1   = (const float*)d_in[12];
    const float* a1_b1   = (const float*)d_in[13];
    const float* a1_W2   = (const float*)d_in[14];
    const float* a1_b2   = (const float*)d_in[15];
    const float* conv1_W = (const float*)d_in[16];
    const float* conv1_b = (const float*)d_in[17];
    const float* conv2_W = (const float*)d_in[18];
    const float* conv2_b = (const float*)d_in[19];
    const float* lin0_W  = (const float*)d_in[20];
    const float* lin0_b  = (const float*)d_in[21];

    const int N = in_sizes[0] / 512;   // IN1 = 512
    const int E = in_sizes[2] / 2;
    const int* row = ei;
    const int* col = ei + E;

    // ---- workspace carve (fp32): H0,H1 [N,256]; T [N,64]; G,S [N,128]; deg,dis [N]
    float* ws  = (float*)d_ws;
    float* H0  = ws;
    float* H1  = H0 + (size_t)N * 256;
    float* T   = H1 + (size_t)N * 256;
    float* G   = T  + (size_t)N * 64;
    float* S   = G  + (size_t)N * 128;
    float* deg = S  + (size_t)N * 128;
    float* dis = deg + N;

    // ---- f16 transposed weight region (64B aligned; all sizes multiple of 32B)
    size_t foff = (size_t)(dis + N - ws);
    foff = (foff + 15) & ~(size_t)15;
    _Float16* hw  = (_Float16*)(ws + foff);
    _Float16* wh0  = hw;                  // [256,256]
    _Float16* wh1  = wh0 + 256 * 256;     // [256,512]
    _Float16* wa01 = wh1 + 512 * 256;     // [64,256]
    _Float16* wa02 = wa01 + 256 * 64;     // [256,64]
    _Float16* wa11 = wa02 + 64 * 256;
    _Float16* wa12 = wa11 + 256 * 64;
    _Float16* wc1  = wa12 + 64 * 256;     // [128,256]
    _Float16* wc2  = wc1 + 256 * 128;     // [128,128]
    _Float16* wl0  = wc2 + 128 * 128;     // [128,128]

    const int TPB = 256;
    auto cdiv = [](long a, long b) { return (int)((a + b - 1) / b); };

    // ---- weight convert+transpose pre-pass (tiny)
    k_w2h_t<<<cdiv(256 * 256, TPB), TPB, 0, stream>>>(hete0_W, wh0, 256, 256);
    k_w2h_t<<<cdiv(512 * 256, TPB), TPB, 0, stream>>>(hete1_W, wh1, 512, 256);
    k_w2h_t<<<cdiv(256 * 64,  TPB), TPB, 0, stream>>>(a0_W1, wa01, 256, 64);
    k_w2h_t<<<cdiv(64 * 256,  TPB), TPB, 0, stream>>>(a0_W2, wa02, 64, 256);
    k_w2h_t<<<cdiv(256 * 64,  TPB), TPB, 0, stream>>>(a1_W1, wa11, 256, 64);
    k_w2h_t<<<cdiv(64 * 256,  TPB), TPB, 0, stream>>>(a1_W2, wa12, 64, 256);
    k_w2h_t<<<cdiv(256 * 128, TPB), TPB, 0, stream>>>(conv1_W, wc1, 256, 128);
    k_w2h_t<<<cdiv(128 * 128, TPB), TPB, 0, stream>>>(conv2_W, wc2, 128, 128);
    k_w2h_t<<<cdiv(128 * 128, TPB), TPB, 0, stream>>>(lin0_W, wl0, 128, 128);

    // ---- degrees / normalization
    k_fill_deg<<<cdiv(N, TPB), TPB, 0, stream>>>(deg, N);
    k_edge_deg<<<cdiv(E, TPB), TPB, 0, stream>>>(col, deg, E);
    k_rsqrt  <<<cdiv(N, TPB), TPB, 0, stream>>>(deg, dis, N);

    const int TM = N / 16;             // 6250 (N % 16 == 0)
    const dim3 B32(32);
    const long nf128 = (long)N * 128, nf256 = (long)N * 256, ef = (long)E * 128;

    // ---- hetero front-end
    wmma_gemm<0, 1, 0, 1><<<dim3(TM, 4), B32, 0, stream>>>(
        x, 512, wh0, hete0_b, nullptr, nullptr, nullptr, types, H0, N, 256, 256);
    wmma_gemm<1, 1, 0, 1><<<dim3(TM, 4), B32, 0, stream>>>(
        x, 512, wh1, hete1_b, nullptr, nullptr, nullptr, types, H1, N, 512, 256);

    // ---- attention MLPs (device-side weight swap when *sidx == 2)
    wmma_gemm<2, 1, 0, 1><<<dim3(TM, 1), B32, 0, stream>>>(
        H0, 256, wa01, a0_b1, wa11, a1_b1, sidx, nullptr, T, N, 256, 64);
    wmma_gemm<2, 0, 1, 1><<<dim3(TM, 4), B32, 0, stream>>>(
        T, 64, wa02, a0_b2, wa12, a1_b2, sidx, nullptr, H0, N, 64, 256);
    wmma_gemm<2, 1, 0, 1><<<dim3(TM, 1), B32, 0, stream>>>(
        H1, 256, wa11, a1_b1, wa01, a0_b1, sidx, nullptr, T, N, 256, 64);
    wmma_gemm<2, 0, 1, 1><<<dim3(TM, 4), B32, 0, stream>>>(
        T, 64, wa12, a1_b2, wa02, a0_b2, sidx, nullptr, H1, N, 64, 256);

    // h = (h0 + a0) + (h1 + a1)
    k_add<<<cdiv(nf256, TPB), TPB, 0, stream>>>(H0, H1, nf256);

    // ---- GCN layer 1
    wmma_gemm<2, 0, 0, 0><<<dim3(TM, 2), B32, 0, stream>>>(
        H0, 256, wc1, nullptr, nullptr, nullptr, nullptr, nullptr, G, N, 256, 128);
    k_selfinit    <<<cdiv(nf128, TPB), TPB, 0, stream>>>(G, dis, S, nf128);
    k_edge_scatter<<<cdiv(ef,    TPB), TPB, 0, stream>>>(row, col, G, dis, S, ef);
    k_bias_relu   <<<cdiv(nf128, TPB), TPB, 0, stream>>>(S, conv1_b, nf128);

    // ---- GCN layer 2
    wmma_gemm<2, 0, 0, 0><<<dim3(TM, 2), B32, 0, stream>>>(
        S, 128, wc2, nullptr, nullptr, nullptr, nullptr, nullptr, G, N, 128, 128);
    k_selfinit    <<<cdiv(nf128, TPB), TPB, 0, stream>>>(G, dis, S, nf128);
    k_edge_scatter<<<cdiv(ef,    TPB), TPB, 0, stream>>>(row, col, G, dis, S, ef);
    k_bias_relu   <<<cdiv(nf128, TPB), TPB, 0, stream>>>(S, conv2_b, nf128);

    // ---- dense head straight into d_out
    wmma_gemm<2, 1, 0, 1><<<dim3(TM, 2), B32, 0, stream>>>(
        S, 128, wl0, lin0_b, nullptr, nullptr, nullptr, nullptr,
        (float*)d_out, N, 128, 128);
}
